// ConvKAN_77120432767745
// MI455X (gfx1250) — matrix operand order repaired and checked by reading, not compiled, not run on hardware
//
#include <hip/hip_runtime.h>
#include <hip/hip_bf16.h>
#include <math.h>

typedef __attribute__((ext_vector_type(16))) _Float16 v16h;
typedef __attribute__((ext_vector_type(8)))  _Float16 v8h;
typedef __attribute__((ext_vector_type(2)))  _Float16 v2h;
typedef __attribute__((ext_vector_type(8)))  float    v8f;

#define IMG 512
#define NT  18          // tiles per side: 18*30 = 540 >= 512
#define OUTT 30         // output interior per tile
#define FEATK 96        // padded K (81 real features -> 96 = 3 * 32)

__device__ __forceinline__ float siluf(float v) {
    return v / (1.0f + __expf(-v));
}

// Uniform cubic B-spline: grid = -2.2 + 0.4*j, j=0..11; 8 cubic bases.
// Only 4 consecutive bases are nonzero; closed-form weights replace the
// Cox-de Boor recursion (~20 flops vs ~160).
__device__ __forceinline__ void spline8(_Float16* dst, float xv) {
    float tp = (xv + 2.2f) * 2.5f;       // (x - g0) / h
    float fm = floorf(tp);
    int   m  = (int)fm;
    if (m < 0 || m > 10) return;         // outside grid -> all-zero bases
    float u  = tp - fm;
    float um = 1.0f - u;
    float u2 = u * u, u3 = u2 * u;
    const float k6 = 1.0f / 6.0f;
    float w0 = um * um * um * k6;
    float w1 = (3.0f * u3 - 6.0f * u2 + 4.0f) * k6;
    float w2 = (-3.0f * u3 + 3.0f * u2 + 3.0f * u + 1.0f) * k6;
    float w3 = u3 * k6;
    int j0 = m - 3;
    if (j0     >= 0 && j0     < 8) dst[j0]     = (_Float16)w0;
    if (j0 + 1 >= 0 && j0 + 1 < 8) dst[j0 + 1] = (_Float16)w1;
    if (j0 + 2 >= 0 && j0 + 2 < 8) dst[j0 + 2] = (_Float16)w2;
    if (j0 + 3 >= 0 && j0 + 3 < 8) dst[j0 + 3] = (_Float16)w3;
}

// Fold base_weight + spline_weight*spline_scaler into the f16 B-matrix (32x16)
// wave32 register layout: frag[(s*32 + lane)*16 + e], lane<16 -> K=s*32+e,
// lane>=16 -> K=s*32+16+e, column (conv channel) = lane & 15.
__global__ __launch_bounds__(256) void convkan_prep(
    const float* __restrict__ bw, const float* __restrict__ sw,
    const float* __restrict__ ss, _Float16* __restrict__ wfrag)
{
    int idx = blockIdx.x * 256 + threadIdx.x;
    if (idx >= 3 * 32 * 16) return;
    int e = idx & 15;
    int l = (idx >> 4) & 31;
    int s = idx >> 9;
    int c = l & 15;
    int k = s * 32 + ((l < 16) ? e : 16 + e);
    float v = 0.0f;
    if (k < 9) {
        v = bw[c * 9 + k];
    } else if (k < 81) {
        int i = (k - 9) >> 3, o = (k - 9) & 7;
        v = sw[(c * 9 + i) * 8 + o] * ss[c * 9 + i];
    }
    wfrag[idx] = (_Float16)v;
}

// One 32x32 feat tile + fused restore conv. INTERIOR=true => every feat pixel
// and every output of this tile is strictly inside the image: no bounds masks.
template <bool INTERIOR>
__device__ __forceinline__ void run_tile(
    const float* __restrict__ xb, const _Float16* __restrict__ wfrag,
    float* __restrict__ out, int b, int oh, int ow,
    float* xs, _Float16* featbuf, _Float16* featOut, float* wconv,
    const float* __restrict__ restore_w, const float* __restrict__ restore_b)
{
    const int t = threadIdx.x;

    // ---- stage 34x34 (zero-padded if border) x tile ----
    for (int i = t; i < 34 * 34; i += 256) {
        int r = i / 34, c = i - r * 34;
        int gh = oh - 1 + r, gw = ow - 1 + c;
        float v;
        if constexpr (INTERIOR) {
            v = xb[gh * IMG + gw];
        } else {
            v = 0.0f;
            if ((unsigned)gh < (unsigned)IMG && (unsigned)gw < (unsigned)IMG)
                v = xb[gh * IMG + gw];
        }
        xs[r * 36 + c] = v;
    }
    if (t < 144) wconv[t] = restore_w[t];
    if (t == 144) wconv[144] = restore_b[0];

    // ---- persistent B fragments (weights), one set per lane ----
    const int lane = t & 31;
    const v16h bf0 = *(const v16h*)(wfrag + (0 * 32 + lane) * 16);
    const v16h bf1 = *(const v16h*)(wfrag + (1 * 32 + lane) * 16);
    const v16h bf2 = *(const v16h*)(wfrag + (2 * 32 + lane) * 16);

    __syncthreads();

    const int wave = t >> 5;
    const int pix  = t & 127;      // pixel-in-chunk for feature phase
    const int hsel = t >> 7;       // which half of the 96 features

    for (int q = 0; q < 8; ++q) {  // 8 chunks of 128 pixels = 32x32 tile
        // zero featbuf (each thread clears 48 f16 = 6 x b128)
        {
            _Float16* zp = featbuf + pix * FEATK + hsel * 48;
            v8h z = {};
            #pragma unroll
            for (int zz = 0; zz < 6; ++zz) ((v8h*)zp)[zz] = z;
        }
        __syncthreads();

        // feature scatter: silu(p[0..8]) -> f[0..8]; bases(p_i) -> f[9+8i..16+8i]
        {
            int p   = q * 128 + pix;
            int row = p >> 5, col = p & 31;
            const float* xc = &xs[row * 36 + col];   // patch top-left
            _Float16* fb = featbuf + pix * FEATK;
            if (hsel == 0) {
                float p00 = xc[0],  p01 = xc[1],  p02 = xc[2];
                float p10 = xc[36], p11 = xc[37], p12 = xc[38];
                float p20 = xc[72], p21 = xc[73], p22 = xc[74];
                // packed b32 stores for the 9 silu features (base is 16B aligned)
                v2h s01 = { (_Float16)siluf(p00), (_Float16)siluf(p01) };
                v2h s23 = { (_Float16)siluf(p02), (_Float16)siluf(p10) };
                v2h s45 = { (_Float16)siluf(p11), (_Float16)siluf(p12) };
                v2h s67 = { (_Float16)siluf(p20), (_Float16)siluf(p21) };
                ((v2h*)fb)[0] = s01; ((v2h*)fb)[1] = s23;
                ((v2h*)fb)[2] = s45; ((v2h*)fb)[3] = s67;
                fb[8] = (_Float16)siluf(p22);
                spline8(fb + 9 + 8 * 0, p00);
                spline8(fb + 9 + 8 * 1, p01);
                spline8(fb + 9 + 8 * 2, p02);
                spline8(fb + 9 + 8 * 3, p10);
            } else {
                float p11 = xc[37], p12 = xc[38];
                float p20 = xc[72], p21 = xc[73], p22 = xc[74];
                spline8(fb + 9 + 8 * 4, p11);
                spline8(fb + 9 + 8 * 5, p12);
                spline8(fb + 9 + 8 * 6, p20);
                spline8(fb + 9 + 8 * 7, p21);
                spline8(fb + 9 + 8 * 8, p22);
            }
        }
        __syncthreads();

        // WMMA: each wave computes one 16-pixel x 16-channel tile, K = 96
        {
            const int tbase = wave * 16;
            const int mrow  = lane & 15;
            const int hi16  = lane >> 4;   // A: elems 0-15 <- K {0..15 | 16..31}
            const _Float16* ap = featbuf + (tbase + mrow) * FEATK + hi16 * 16;
            v8f acc = {};
            #pragma unroll
            for (int s = 0; s < 3; ++s) {
                v8h alo = *(const v8h*)(ap + s * 32);        // K base + {0|16}
                v8h ahi = *(const v8h*)(ap + s * 32 + 8);    // K base + 8 + {0|16}
                v16h a;
                #pragma unroll
                for (int e = 0; e < 8; ++e) { a[e] = alo[e]; a[8 + e] = ahi[e]; }
                v16h bsel = (s == 0) ? bf0 : ((s == 1) ? bf1 : bf2);
                acc = __builtin_amdgcn_wmma_f32_16x16x32_f16(
                    false, a, false, bsel, (short)0, acc, false, false);
            }
            // C layout: lane -> channel (lane&15), VGPR r -> pixel r (+8 if lane>=16)
            const int ch = lane & 15;
            #pragma unroll
            for (int r = 0; r < 8; ++r) {
                int m = r + hi16 * 8;
                int p = q * 128 + tbase + m;
                if constexpr (INTERIOR) {
                    featOut[p * 16 + ch] = (_Float16)acc[r];
                } else {
                    int row = p >> 5, col = p & 31;
                    int gh  = oh + row, gw = ow + col;
                    bool in = ((unsigned)gh < (unsigned)IMG) &&
                              ((unsigned)gw < (unsigned)IMG);
                    float v = in ? acc[r] : 0.0f;  // feat zero-padded for restore conv
                    featOut[p * 16 + ch] = (_Float16)v;
                }
            }
        }
        __syncthreads();
    }

    // ---- fused restore conv (16 -> 1, 3x3) on the interior 30x30 ----
    const float bias = wconv[144];
    for (int o = t; o < OUTT * OUTT; o += 256) {
        int orow = o / OUTT + 1, ocol = o - (orow - 1) * OUTT + 1;
        int gh = oh + orow, gw = ow + ocol;
        if constexpr (!INTERIOR) {
            if ((unsigned)gh >= (unsigned)IMG || (unsigned)gw >= (unsigned)IMG)
                continue;
        }
        float acc = bias;
        #pragma unroll
        for (int dr = 0; dr < 3; ++dr) {
            #pragma unroll
            for (int dc = 0; dc < 3; ++dc) {
                const _Float16* fp =
                    featOut + (((orow - 1 + dr) * 32) + (ocol - 1 + dc)) * 16;
                v8h f0 = *(const v8h*)fp;
                v8h f1 = *(const v8h*)(fp + 8);
                #pragma unroll
                for (int ch = 0; ch < 8; ++ch) {
                    acc += (float)f0[ch] * wconv[(ch * 3 + dr) * 3 + dc];
                    acc += (float)f1[ch] * wconv[((ch + 8) * 3 + dr) * 3 + dc];
                }
            }
        }
        out[(b * IMG + gh) * IMG + gw] = acc;
    }
}

__global__ __launch_bounds__(256) void convkan_main(
    const float* __restrict__ x, const float* __restrict__ restore_w,
    const float* __restrict__ restore_b, const _Float16* __restrict__ wfrag,
    float* __restrict__ out)
{
    __shared__ __align__(16) float    xs[34 * 36];          //  4896 B (x halo tile)
    __shared__ __align__(16) _Float16 featbuf[128 * FEATK]; // 24576 B (A staging)
    __shared__ __align__(16) _Float16 featOut[1024 * 16];   // 32768 B (feat tile)
    __shared__ __align__(16) float    wconv[160];           //   640 B

    const int tile = blockIdx.x;
    const int b    = tile / (NT * NT);
    const int rem  = tile - b * NT * NT;
    const int tr   = rem / NT, tc = rem - tr * NT;
    const int oh   = tr * OUTT - 1;                 // feat-tile origin (row)
    const int ow   = tc * OUTT - 1;                 // feat-tile origin (col)
    const float* xb = x + b * IMG * IMG;

    // Fully-interior tiles (79%): x halo [oh-1..oh+32]x[ow-1..ow+32] and all
    // 30x30 outputs are in-bounds -> no masks anywhere (block-uniform branch).
    const bool interior = (tr >= 1) && (tr <= 16) && (tc >= 1) && (tc <= 16);
    if (interior)
        run_tile<true >(xb, wfrag, out, b, oh, ow, xs, featbuf, featOut, wconv,
                        restore_w, restore_b);
    else
        run_tile<false>(xb, wfrag, out, b, oh, ow, xs, featbuf, featOut, wconv,
                        restore_w, restore_b);
}

extern "C" void kernel_launch(void* const* d_in, const int* in_sizes, int n_in,
                              void* d_out, int out_size, void* d_ws, size_t ws_size,
                              hipStream_t stream) {
    (void)in_sizes; (void)n_in; (void)out_size; (void)ws_size;
    const float* x             = (const float*)d_in[0];
    const float* base_weight   = (const float*)d_in[1];
    const float* spline_weight = (const float*)d_in[2];
    const float* spline_scaler = (const float*)d_in[3];
    const float* restore_w     = (const float*)d_in[4];
    const float* restore_b     = (const float*)d_in[5];
    float*       out           = (float*)d_out;
    _Float16*    wfrag         = (_Float16*)d_ws;   // 1536 f16 = 3 KB

    convkan_prep<<<6, 256, 0, stream>>>(base_weight, spline_weight,
                                        spline_scaler, wfrag);
    convkan_main<<<8 * NT * NT, 256, 0, stream>>>(x, restore_w, restore_b,
                                                  wfrag, out);
}
